// MG_25031069401696
// MI455X (gfx1250) — compile-verified, matrix operand order checked by reading, not run on hardware
//
#include <hip/hip_runtime.h>

// ---------------- problem constants (from reference) ----------------
#define N_NODES 8192
#define HF      512        // F == H == 512
#define NL      2

// GEMM tiling: block = 256 threads = 8 wave32s, tile 128x64, K-step 32
#define BM 128
#define BN 64
#define BK 32

typedef __attribute__((ext_vector_type(16))) __bf16 v16bf;
typedef __attribute__((ext_vector_type(8)))  float  v8f;
typedef __attribute__((ext_vector_type(4)))  unsigned v4u;
typedef __attribute__((ext_vector_type(8)))  unsigned v8u;

struct U16x16 { unsigned short u[16]; };

__device__ __forceinline__ unsigned short f2bf(float f) {
    unsigned u = __builtin_bit_cast(unsigned, f);
    u += 0x7FFFu + ((u >> 16) & 1u);       // round-to-nearest-even bf16
    return (unsigned short)(u >> 16);
}

// CDNA5 16-bit A/B fragment K index for element j, K-half 'hi' (lane>>4)
__device__ __forceinline__ int wmma_k_of(int j, int hi) {
    int v = j >> 1, p = j & 1;
    return ((v < 4) ? (v * 2 + p) : (16 + (v - 4) * 2 + p)) + hi * 8;
}

// ---------------- TDM: DMA one bf16 tile (tile_rows x 32) of a row-major
// 8192x512 bf16 tensor from global into LDS (rows stored contiguously). ----------------
__device__ __forceinline__ void tdm_load_tile_bf16(unsigned lds_addr,
                                                   const unsigned short* gptr,
                                                   unsigned tile_rows) {
    unsigned long long ga = (unsigned long long)(size_t)gptr;
    // D# group 0: count=1 | lds_addr | global_addr[56:0] | type=2
    v4u g0 = { 1u,
               lds_addr,
               (unsigned)ga,
               (unsigned)((ga >> 32) & 0x01FFFFFFu) | (2u << 30) };
    // D# group 1: data_size=2B; tensor_dim0=512, tensor_dim1=8192;
    // tile_dim0=32, tile_dim1=tile_rows; tensor_dim0_stride=512
    v8u g1 = { 1u << 16,          // wg_mask=0, data_size=1 (2 bytes)
               512u << 16,        // tensor_dim0[15:0] in bits 63:48
               8192u << 16,       // tensor_dim0[31:16]=0 | tensor_dim1[15:0]
               32u << 16,         // tensor_dim1[31:16]=0 | tile_dim0=32
               tile_rows,         // tile_dim1 | tile_dim2=0
               512u,              // tensor_dim0_stride[31:0]
               0u, 0u };
    asm volatile("tensor_load_to_lds %0, %1" :: "s"(g0), "s"(g1) : "memory");
}

// ---------------- small utility kernels ----------------
__global__ void copy4_kernel(const float4* __restrict__ a, float4* __restrict__ o, int n4) {
    int i = blockIdx.x * blockDim.x + threadIdx.x;
    if (i < n4) o[i] = a[i];
}

__global__ __launch_bounds__(128) void token_set_kernel(float* __restrict__ x,
                                                        const int* __restrict__ tok,
                                                        const float* __restrict__ mask) {
    int node = tok[blockIdx.x];
    ((float4*)(x + (size_t)node * HF))[threadIdx.x] = ((const float4*)mask)[threadIdx.x];
}

__global__ __launch_bounds__(128) void noise_set_kernel(float* __restrict__ x,
                                                        const int* __restrict__ nn,
                                                        const int* __restrict__ ns,
                                                        const float* __restrict__ attr) {
    int dnode = nn[blockIdx.x];
    int snode = ns[blockIdx.x];
    ((float4*)(x + (size_t)dnode * HF))[threadIdx.x] =
        ((const float4*)(attr + (size_t)snode * HF))[threadIdx.x];
}

__global__ void deg_kernel(const int* __restrict__ idx, int E, float* __restrict__ deg) {
    int i = blockIdx.x * blockDim.x + threadIdx.x;
    if (i < E) atomicAdd(deg + idx[i], 1.0f);
}

__global__ void dinv_kernel(float* __restrict__ d, int n) {
    int i = blockIdx.x * blockDim.x + threadIdx.x;
    if (i < n) d[i] = rsqrtf(fmaxf(d[i], 1.0f));
}

// ---------------- message passing: m[dst] += h[src] * (dsrc[src]*ddst[dst]) ----------------
__global__ __launch_bounds__(128) void msgpass_kernel(const float* __restrict__ h,
                                                      const int* __restrict__ src,
                                                      const int* __restrict__ dst,
                                                      int E,
                                                      const float* __restrict__ dsrc,
                                                      const float* __restrict__ ddst,
                                                      int scaled,
                                                      float* __restrict__ m) {
    int e = blockIdx.x;
    if (e >= E) return;
    int s = src[e], d = dst[e];
    float scale = scaled ? (dsrc[s] * ddst[d]) : 1.0f;
    float4 v = ((const float4*)(h + (size_t)s * HF))[threadIdx.x];
    float* md = m + (size_t)d * HF + threadIdx.x * 4;
    atomicAdd(md + 0, v.x * scale);
    atomicAdd(md + 1, v.y * scale);
    atomicAdd(md + 2, v.z * scale);
    atomicAdd(md + 3, v.w * scale);
}

// ---------------- WMMA GEMM:  t = prelu(A(MxK) @ W(KxNn) + bias, ain[layer]) ----------------
__global__ __launch_bounds__(256) void gemm_bias_prelu(const float* __restrict__ A,
                                                       const float* __restrict__ W,
                                                       const float* __restrict__ bias,
                                                       const float* __restrict__ ain, int layer,
                                                       float* __restrict__ C,
                                                       int M, int K, int Nn) {
    __shared__ unsigned short sA [BM][BK + 2];
    __shared__ unsigned short sBt[BN][BK + 2];   // n-major (transposed) -> paired ds loads
    const int tid  = threadIdx.x;
    const int lane = tid & 31;
    const int wave = tid >> 5;
    const int mBlock = blockIdx.y * BM;
    const int nBlock = blockIdx.x * BN;
    const int wm = (wave >> 1) * 32;
    const int wn = (wave & 1) * 32;
    const int rr = lane & 15;
    const int hi = lane >> 4;

    v8f acc[2][2] = {};

    for (int k0 = 0; k0 < K; k0 += BK) {
        // stage A (fp32 -> bf16) into LDS
        #pragma unroll
        for (int i = 0; i < (BM * BK) / 256; ++i) {
            int idx = tid + i * 256;
            int r = idx >> 5, c = idx & 31;
            sA[r][c] = f2bf(A[(size_t)(mBlock + r) * K + (k0 + c)]);
        }
        // stage W (fp32 -> bf16) transposed into LDS (coalesced global reads)
        #pragma unroll
        for (int i = 0; i < (BK * BN) / 256; ++i) {
            int idx = tid + i * 256;
            int r = idx >> 6, c = idx & 63;
            sBt[c][r] = f2bf(W[(size_t)(k0 + r) * Nn + (nBlock + c)]);
        }
        __syncthreads();

        #pragma unroll
        for (int mi = 0; mi < 2; ++mi) {
            U16x16 af;
            #pragma unroll
            for (int j = 0; j < 16; ++j)
                af.u[j] = sA[wm + mi * 16 + rr][wmma_k_of(j, hi)];
            v16bf a = __builtin_bit_cast(v16bf, af);
            #pragma unroll
            for (int ni = 0; ni < 2; ++ni) {
                U16x16 bfm;
                #pragma unroll
                for (int j = 0; j < 16; ++j)
                    bfm.u[j] = sBt[wn + ni * 16 + rr][wmma_k_of(j, hi)];
                v16bf b = __builtin_bit_cast(v16bf, bfm);
                acc[mi][ni] = __builtin_amdgcn_wmma_f32_16x16x32_bf16(
                    false, a, false, b, (short)0, acc[mi][ni], false, false);
            }
        }
        __syncthreads();
    }

    const float slope = ain[layer];
    #pragma unroll
    for (int mi = 0; mi < 2; ++mi) {
        #pragma unroll
        for (int ni = 0; ni < 2; ++ni) {
            #pragma unroll
            for (int r = 0; r < 8; ++r) {
                int mrow = mBlock + wm + mi * 16 + r + hi * 8;
                int ncol = nBlock + wn + ni * 16 + rr;
                float v = acc[mi][ni][r] + bias[ncol];
                v = (v >= 0.0f) ? v : slope * v;
                C[(size_t)mrow * Nn + ncol] = v;
            }
        }
    }
}

// ---------------- batch-norm stats over rows (per column) ----------------
__global__ __launch_bounds__(256) void bnstats_kernel(const float* __restrict__ t,
                                                      float* __restrict__ meanv,
                                                      float* __restrict__ invstd,
                                                      int M, int Nn) {
    __shared__ float ssum[4][64];
    __shared__ float ssq[4][64];
    int c  = threadIdx.x & 63;
    int rg = threadIdx.x >> 6;
    int col = blockIdx.x * 64 + c;
    float s = 0.0f, q = 0.0f;
    for (int r = rg; r < M; r += 4) {
        float v = t[(size_t)r * Nn + col];
        s += v; q += v * v;
    }
    ssum[rg][c] = s; ssq[rg][c] = q;
    __syncthreads();
    if (rg == 0) {
        s = ssum[0][c] + ssum[1][c] + ssum[2][c] + ssum[3][c];
        q = ssq[0][c] + ssq[1][c] + ssq[2][c] + ssq[3][c];
        float mu  = s / (float)M;
        float var = q / (float)M - mu * mu;
        meanv[col]  = mu;
        invstd[col] = rsqrtf(var + 1e-5f);
    }
}

// ---------------- bn apply + prelu(aout) ----------------
__global__ __launch_bounds__(256) void bnapply_kernel(const float* __restrict__ t,
                                                      const float* __restrict__ meanv,
                                                      const float* __restrict__ invstd,
                                                      const float* __restrict__ g,
                                                      const float* __restrict__ bb,
                                                      const float* __restrict__ aout, int layer,
                                                      float* __restrict__ h) {
    const float slope = aout[layer];
    int base = (blockIdx.x * 256 + threadIdx.x) * 4;
    #pragma unroll
    for (int k = 0; k < 4; ++k) {
        int i = base + k;
        int c = i & (HF - 1);
        float v = (t[i] - meanv[c]) * invstd[c] * g[c] + bb[c];
        h[i] = (v >= 0.0f) ? v : slope * v;
    }
}

// ---------------- per-row L2 normalize, emitting bf16 (feeds TDM/WMMA SYRK) ----------------
__global__ __launch_bounds__(256) void l2row_bf16_kernel(const float* __restrict__ h,
                                                         unsigned short* __restrict__ o) {
    __shared__ float red[256];
    __shared__ float norm;
    int row = blockIdx.x;
    const float* hr = h + (size_t)row * HF;
    float q = 0.0f;
    for (int i = threadIdx.x; i < HF; i += 256) { float v = hr[i]; q += v * v; }
    red[threadIdx.x] = q;
    __syncthreads();
    for (int s = 128; s > 0; s >>= 1) {
        if (threadIdx.x < s) red[threadIdx.x] += red[threadIdx.x + s];
        __syncthreads();
    }
    if (threadIdx.x == 0) norm = fmaxf(sqrtf(red[0]), 1e-12f);
    __syncthreads();
    float inv = 1.0f / norm;
    for (int i = threadIdx.x; i < HF; i += 256)
        o[(size_t)row * HF + i] = f2bf(hr[i] * inv);
}

// ---------------- loss1: mean over nodes of (1 - cos(h1, attr))^3 ----------------
__global__ __launch_bounds__(256) void loss1_kernel(const float* __restrict__ h1,
                                                    const float* __restrict__ attr,
                                                    float* __restrict__ acc) {
    __shared__ float sd[256], sa[256], sy[256];
    int row = blockIdx.x;
    const float* a = h1 + (size_t)row * HF;
    const float* y = attr + (size_t)row * HF;
    float dot = 0.0f, na = 0.0f, ny = 0.0f;
    for (int i = threadIdx.x; i < HF; i += 256) {
        float va = a[i], vy = y[i];
        dot += va * vy; na += va * va; ny += vy * vy;
    }
    sd[threadIdx.x] = dot; sa[threadIdx.x] = na; sy[threadIdx.x] = ny;
    __syncthreads();
    for (int s = 128; s > 0; s >>= 1) {
        if (threadIdx.x < s) {
            sd[threadIdx.x] += sd[threadIdx.x + s];
            sa[threadIdx.x] += sa[threadIdx.x + s];
            sy[threadIdx.x] += sy[threadIdx.x + s];
        }
        __syncthreads();
    }
    if (threadIdx.x == 0) {
        float cosv = sd[0] / (fmaxf(sqrtf(sa[0]), 1e-12f) * fmaxf(sqrtf(sy[0]), 1e-12f));
        float c = 1.0f - cosv;
        atomicAdd(acc, c * c * c);
    }
}

// ---------------- fused SYRK + MSE, TDM double-buffered:
//   sum over tile of (h2n@h2n^T - matrix)^2, new_matrix never materialized. ----------------
__global__ __launch_bounds__(256) void syrk_loss_tdm_kernel(const unsigned short* __restrict__ Hb,
                                                            const float* __restrict__ Mtx,
                                                            float* __restrict__ acc) {
    // TDM writes tile rows contiguously -> unpadded [rows][32] bf16 buffers, 2x for ping-pong
    __shared__ unsigned short sA[2][BM][BK];
    __shared__ unsigned short sB[2][BN][BK];   // n-major: row n holds K-slice of Hn[jBlock+n]
    __shared__ float red[256];
    const int tid  = threadIdx.x;
    const int lane = tid & 31;
    const int wave = tid >> 5;
    const int iBlock = blockIdx.y * BM;
    const int jBlock = blockIdx.x * BN;
    const int wm = (wave >> 1) * 32;
    const int wn = (wave & 1) * 32;
    const int rr = lane & 15;
    const int hi = lane >> 4;

    // warm GL2 with the matrix tile needed in the epilogue
    __builtin_prefetch(Mtx + (size_t)(iBlock + (tid >> 1)) * N_NODES + jBlock + (tid & 1) * 32, 0, 1);

    const unsigned short* Arow = Hb + (size_t)iBlock * HF;
    const unsigned short* Brow = Hb + (size_t)jBlock * HF;
    const unsigned ldsA = (unsigned)(size_t)(&sA[0][0][0]);
    const unsigned ldsB = (unsigned)(size_t)(&sB[0][0][0]);

    v8f acc4[2][2] = {};
    int buf = 0;

    // prologue: DMA first K-slice (wave 0 drives the TDM; EXEC is ignored by tensor ops)
    if (tid < 32) {
        tdm_load_tile_bf16(ldsA, Arow, BM);
        tdm_load_tile_bf16(ldsB, Brow, BN);
    }

    for (int k0 = 0; k0 < HF; k0 += BK) {
        if (tid < 32) __builtin_amdgcn_s_wait_tensorcnt(0);
        __syncthreads();                       // current buffer ready for all waves
        int nbuf = buf ^ 1;
        if ((k0 + BK < HF) && (tid < 32)) {    // overlap next DMA with this tile's WMMAs
            tdm_load_tile_bf16(ldsA + (unsigned)(nbuf * BM * BK * 2), Arow + (k0 + BK), BM);
            tdm_load_tile_bf16(ldsB + (unsigned)(nbuf * BN * BK * 2), Brow + (k0 + BK), BN);
        }

        #pragma unroll
        for (int mi = 0; mi < 2; ++mi) {
            U16x16 af;
            #pragma unroll
            for (int j = 0; j < 16; ++j)
                af.u[j] = sA[buf][wm + mi * 16 + rr][wmma_k_of(j, hi)];
            v16bf a = __builtin_bit_cast(v16bf, af);
            #pragma unroll
            for (int ni = 0; ni < 2; ++ni) {
                U16x16 bfm;
                #pragma unroll
                for (int j = 0; j < 16; ++j)
                    bfm.u[j] = sB[buf][wn + ni * 16 + rr][wmma_k_of(j, hi)];
                v16bf b = __builtin_bit_cast(v16bf, bfm);
                acc4[mi][ni] = __builtin_amdgcn_wmma_f32_16x16x32_bf16(
                    false, a, false, b, (short)0, acc4[mi][ni], false, false);
            }
        }
        __syncthreads();                       // done reading 'buf' before TDM reuses it
        buf = nbuf;
    }

    float lsum = 0.0f;
    #pragma unroll
    for (int mi = 0; mi < 2; ++mi) {
        #pragma unroll
        for (int ni = 0; ni < 2; ++ni) {
            #pragma unroll
            for (int r = 0; r < 8; ++r) {
                int irow = iBlock + wm + mi * 16 + r + hi * 8;
                int jcol = jBlock + wn + ni * 16 + rr;
                float d = acc4[mi][ni][r] - Mtx[(size_t)irow * N_NODES + jcol];
                lsum += d * d;
            }
        }
    }
    red[tid] = lsum;
    __syncthreads();
    for (int s = 128; s > 0; s >>= 1) {
        if (tid < s) red[tid] += red[tid + s];
        __syncthreads();
    }
    if (tid == 0) atomicAdd(acc, red[0]);
}

__global__ void finalize_kernel(const float* __restrict__ acc, float* __restrict__ out) {
    float l1 = acc[0] / (float)N_NODES;
    float l2 = acc[1] / ((float)N_NODES * (float)N_NODES);
    out[0] = 0.5f * l1 + 0.5f * l2;
}

// ---------------- host orchestration ----------------
static void run_net(const float* hin, const int* src, const int* dst, int E,
                    const float* dsrc, const float* ddst, int scaled,
                    const float* W, const float* b, const float* ain,
                    const float* g, const float* bb, const float* aout,
                    float* m, float* t, float* meanv, float* invsd,
                    float* hmid, float* hout, hipStream_t stream) {
    const float* cur = hin;
    float* dsts[2] = { hmid, hout };
    for (int i = 0; i < NL; ++i) {
        hipMemsetAsync(m, 0, (size_t)N_NODES * HF * sizeof(float), stream);
        msgpass_kernel<<<E, 128, 0, stream>>>(cur, src, dst, E, dsrc, ddst, scaled, m);
        dim3 gg(HF / BN, N_NODES / BM);
        gemm_bias_prelu<<<gg, 256, 0, stream>>>(m, W + (size_t)i * HF * HF,
                                                b + (size_t)i * HF, ain, i, t,
                                                N_NODES, HF, HF);
        bnstats_kernel<<<HF / 64, 256, 0, stream>>>(t, meanv, invsd, N_NODES, HF);
        bnapply_kernel<<<(N_NODES * HF) / 1024, 256, 0, stream>>>(
            t, meanv, invsd, g + (size_t)i * HF, bb + (size_t)i * HF, aout, i, dsts[i]);
        cur = dsts[i];
    }
}

extern "C" void kernel_launch(void* const* d_in, const int* in_sizes, int n_in,
                              void* d_out, int out_size, void* d_ws, size_t ws_size,
                              hipStream_t stream) {
    const float* attr     = (const float*)d_in[0];
    const float* matrix   = (const float*)d_in[1];
    const float* mask_tok = (const float*)d_in[2];
    const int*   src      = (const int*)d_in[3];
    const int*   dst      = (const int*)d_in[4];
    const int*   src2     = (const int*)d_in[5];
    const int*   dst2     = (const int*)d_in[6];
    const int*   tok      = (const int*)d_in[7];
    const int*   nnodes   = (const int*)d_in[8];
    const int*   nsrc     = (const int*)d_in[9];
    const float* encW = (const float*)d_in[10]; const float* encB = (const float*)d_in[11];
    const float* encAin = (const float*)d_in[12]; const float* encG = (const float*)d_in[13];
    const float* encBB = (const float*)d_in[14]; const float* encAout = (const float*)d_in[15];
    const float* d1W = (const float*)d_in[16]; const float* d1B = (const float*)d_in[17];
    const float* d1Ain = (const float*)d_in[18]; const float* d1G = (const float*)d_in[19];
    const float* d1BB = (const float*)d_in[20]; const float* d1Aout = (const float*)d_in[21];
    const float* d2W = (const float*)d_in[22]; const float* d2B = (const float*)d_in[23];
    const float* d2Ain = (const float*)d_in[24]; const float* d2G = (const float*)d_in[25];
    const float* d2BB = (const float*)d_in[26]; const float* d2Aout = (const float*)d_in[27];

    const int E1  = in_sizes[3];
    const int E2v = in_sizes[5];
    const int NT  = in_sizes[7];
    const int NNO = in_sizes[8];

    float* ws = (float*)d_ws;
    const size_t NH = (size_t)N_NODES * HF;
    float* x     = ws;
    float* m     = ws + 1 * NH;
    float* t     = ws + 2 * NH;
    float* hA    = ws + 3 * NH;
    float* hB    = ws + 4 * NH;
    float* h1    = ws + 5 * NH;
    unsigned short* h2nb = (unsigned short*)(ws + 6 * NH);   // bf16, NH elements = NH/2 floats
    float* deg1s = ws + 6 * NH + NH / 2;
    float* deg1d = deg1s + N_NODES;
    float* deg2s = deg1d + N_NODES;
    float* deg2d = deg2s + N_NODES;
    float* meanv = deg2d + N_NODES;
    float* invsd = meanv + HF;
    float* lossA = invsd + HF;

    // degrees -> rsqrt(clip(deg,1))
    hipMemsetAsync(deg1s, 0, 4 * N_NODES * sizeof(float), stream);
    deg_kernel<<<(E1 + 255) / 256, 256, 0, stream>>>(src, E1, deg1s);
    deg_kernel<<<(E1 + 255) / 256, 256, 0, stream>>>(dst, E1, deg1d);
    deg_kernel<<<(E2v + 255) / 256, 256, 0, stream>>>(src2, E2v, deg2s);
    deg_kernel<<<(E2v + 255) / 256, 256, 0, stream>>>(dst2, E2v, deg2d);
    dinv_kernel<<<(4 * N_NODES + 255) / 256, 256, 0, stream>>>(deg1s, 4 * N_NODES);

    // x = attr; x[token]=mask; x[noise]=attr[noise_src]  (index sets are disjoint)
    copy4_kernel<<<(N_NODES * HF / 4 + 255) / 256, 256, 0, stream>>>(
        (const float4*)attr, (float4*)x, N_NODES * HF / 4);
    token_set_kernel<<<NT, 128, 0, stream>>>(x, tok, mask_tok);
    noise_set_kernel<<<NNO, 128, 0, stream>>>(x, nnodes, nsrc, attr);

    // h1 = enc(x, graph1) ; h2 = enc(attr, graph2)
    run_net(x,    src,  dst,  E1,  deg1s, deg1d, 1, encW, encB, encAin, encG, encBB, encAout,
            m, t, meanv, invsd, hA, h1, stream);
    run_net(attr, src2, dst2, E2v, deg2s, deg2d, 1, encW, encB, encAin, encG, encBB, encAout,
            m, t, meanv, invsd, hA, hB, stream);
    // h1 = dec1(h1, graph1) ; h2 = dec2(h2, graph2)
    run_net(h1, src,  dst,  E1,  nullptr, nullptr, 0, d1W, d1B, d1Ain, d1G, d1BB, d1Aout,
            m, t, meanv, invsd, hA, h1, stream);
    run_net(hB, src2, dst2, E2v, nullptr, nullptr, 0, d2W, d2B, d2Ain, d2G, d2BB, d2Aout,
            m, t, meanv, invsd, hA, hB, stream);

    // h2n = l2norm(h2) -> bf16 (feeds TDM-driven SYRK)
    l2row_bf16_kernel<<<N_NODES, 256, 0, stream>>>(hB, h2nb);

    hipMemsetAsync(lossA, 0, 2 * sizeof(float), stream);
    loss1_kernel<<<N_NODES, 256, 0, stream>>>(h1, attr, lossA);
    syrk_loss_tdm_kernel<<<dim3(N_NODES / BN, N_NODES / BM), 256, 0, stream>>>(h2nb, matrix, lossA + 1);
    finalize_kernel<<<1, 1, 0, stream>>>(lossA, (float*)d_out);
}
